// VectorQuantizer_52828097741017
// MI455X (gfx1250) — matrix operand order repaired and checked by reading, not compiled, not run on hardware
//
#include <hip/hip_runtime.h>
#include <hip/hip_bf16.h>

typedef __bf16 bf16_t;
typedef __attribute__((ext_vector_type(16))) __bf16 v16bf;
typedef __attribute__((ext_vector_type(8)))  __bf16 v8bf;
typedef __attribute__((ext_vector_type(8)))  float  v8f;

#define NUM_EMB 8192
#define EMB_DIM 512
#define SPATIAL 8192   // 8*32*32
#define BATCH   2
#define NVEC    (BATCH * SPATIAL)  // 16384

#define KT       64                // codebook rows staged per LDS buffer
#define LDS_ROW  1040              // 512*2 bytes + 16B pad (bank-conflict-free b128 reads)
#define BUF_BYTES ((size_t)KT * LDS_ROW)

// ---------------------------------------------------------------------------
// Prep: z [B,C,S] f32  ->  zbf [N=B*S, C] bf16 (tiled transpose via LDS)
// ---------------------------------------------------------------------------
__global__ void k_zprep(const float* __restrict__ z, bf16_t* __restrict__ zbf) {
    __shared__ float tile[32][33];
    const int s0 = blockIdx.x * 32;
    const int c0 = blockIdx.y * 32;
    const int b  = blockIdx.z;
    const int tx = threadIdx.x, ty = threadIdx.y;
    const float* zb = z + (size_t)b * EMB_DIM * SPATIAL;
#pragma unroll
    for (int j = 0; j < 4; ++j) {
        int cl = ty + 8 * j;
        tile[cl][tx] = zb[(size_t)(c0 + cl) * SPATIAL + s0 + tx];  // coalesced in s
    }
    __syncthreads();
#pragma unroll
    for (int j = 0; j < 4; ++j) {
        int sl = ty + 8 * j;
        int n  = b * SPATIAL + s0 + sl;
        zbf[(size_t)n * EMB_DIM + c0 + tx] = (bf16_t)tile[tx][sl]; // coalesced in c
    }
}

// ---------------------------------------------------------------------------
// Prep: emb [K,C] f32 -> ebf [K,C] bf16, enorm[K] = |e|^2 (f32)
// ---------------------------------------------------------------------------
__global__ void k_eprep(const float* __restrict__ emb, bf16_t* __restrict__ ebf,
                        float* __restrict__ enorm) {
    const int r = blockIdx.x;
    const int t = threadIdx.x;
    float v0 = emb[(size_t)r * EMB_DIM + t];
    float v1 = emb[(size_t)r * EMB_DIM + 256 + t];
    ebf[(size_t)r * EMB_DIM + t]       = (bf16_t)v0;
    ebf[(size_t)r * EMB_DIM + 256 + t] = (bf16_t)v1;
    __shared__ float red[256];
    red[t] = v0 * v0 + v1 * v1;
    __syncthreads();
    for (int off = 128; off > 0; off >>= 1) {
        if (t < off) red[t] += red[t + off];
        __syncthreads();
    }
    if (t == 0) enorm[r] = red[0];
}

// ---------------------------------------------------------------------------
// Main: fused GEMM (z_flat . emb^T) + argmin_k( |e_k|^2 - 2 z.e_k )
//   block = 256 threads = 8 waves (2 waves/SIMD). Each wave owns 32 z-rows
//   (two 16-row M-subtiles) with the full A operand (32x512 bf16) resident in
//   256 VGPRs. Codebook tiles are double-buffered through LDS via gfx1250
//   async global->LDS copies (ASYNCcnt) overlapped with compute.
//   Per half-stage the wave runs TWO B streams (two 16-code tiles) x TWO
//   M-subtiles = 4 independent WMMA accumulation chains, so each ds wait
//   covers four in-flight WMMAs and both live B fragments get distinct regs.
// ---------------------------------------------------------------------------
__global__ void __launch_bounds__(256, 2)
k_argmin(const bf16_t* __restrict__ zbf,
         const bf16_t* __restrict__ ebf,
         const float* __restrict__ enorm,
         int* __restrict__ idx_out) {
    extern __shared__ char smem[];
    const int tid  = threadIdx.x;
    const int lane = tid & 31;
    const int wave = tid >> 5;
    const int h    = lane >> 4;   // lane half (selects K sub-range in frags)
    const int nl   = lane & 15;   // M for A-frag, N for B/C/D-frags
    const int row_base = blockIdx.x * 256 + wave * 32;

    // --- A operand: 2 M-subtiles x 16 rows x 512 cols bf16 in 256 VGPRs.
    // ISA 16-bit A 16x32 layout: lane holds row M=nl; element e -> K =
    // (e&7) + 8*h + 16*(e>>3)  => two contiguous 16B chunks per fragment.
    v16bf a0[16], a1[16];
    {
        const bf16_t* zrow0 = zbf + (size_t)(row_base + nl) * EMB_DIM;
        const bf16_t* zrow1 = zbf + (size_t)(row_base + 16 + nl) * EMB_DIM;
#pragma unroll
        for (int cc = 0; cc < 16; ++cc) {
            const int cb = cc * 32;
            v8bf lo0 = *(const v8bf*)(zrow0 + cb + 8 * h);
            v8bf hi0 = *(const v8bf*)(zrow0 + cb + 16 + 8 * h);
            a0[cc] = __builtin_shufflevector(lo0, hi0, 0, 1, 2, 3, 4, 5, 6, 7,
                                             8, 9, 10, 11, 12, 13, 14, 15);
            v8bf lo1 = *(const v8bf*)(zrow1 + cb + 8 * h);
            v8bf hi1 = *(const v8bf*)(zrow1 + cb + 16 + 8 * h);
            a1[cc] = __builtin_shufflevector(lo1, hi1, 0, 1, 2, 3, 4, 5, 6, 7,
                                             8, 9, 10, 11, 12, 13, 14, 15);
        }
    }

    float mind0[8], mind1[8];
    int   midx0[8], midx1[8];
#pragma unroll
    for (int i = 0; i < 8; ++i) {
        mind0[i] = 3.4e38f; midx0[i] = 0;
        mind1[i] = 3.4e38f; midx1[i] = 0;
    }

    // Base LDS byte offset of the dynamic-LDS buffer (flat addr low 32 bits).
    const uint32_t lds_base = (uint32_t)(uintptr_t)(void*)smem;

    // Cooperative async stage of KT codebook rows (KT*1024 B) into LDS.
    // Per thread: 16 x b128 chunks; fixed 16B column, rows r0+4i (padded LDS rows).
    auto stage_async = [&](int ktile, uint32_t dst_lds) {
        const unsigned long long sbase =
            (unsigned long long)(uintptr_t)(ebf + (size_t)ktile * KT * EMB_DIM);
        const int c16 = tid & 63;
        const int r0  = tid >> 6;
        uint32_t voff = (uint32_t)(r0 * 1024 + c16 * 16);                 // global byte off
        uint32_t loff = dst_lds + (uint32_t)(r0 * LDS_ROW + c16 * 16);    // LDS byte addr
#pragma unroll
        for (int i = 0; i < 16; ++i) {
            asm volatile("global_load_async_to_lds_b128 %0, %1, %2"
                         :: "v"(loff), "v"(voff), "s"(sbase)
                         : "memory");
            voff += 4096u;            // 4 rows * 1024 B in global
            loff += 4u * LDS_ROW;     // 4 rows * 1040 B in LDS
        }
    };

    const int nstage = NUM_EMB / KT;  // 128
    stage_async(0, lds_base);
    asm volatile("s_wait_asynccnt 0x0" ::: "memory");
    __syncthreads();

    for (int kt = 0; kt < nstage; ++kt) {
        const char* cur = smem + (size_t)(kt & 1) * BUF_BYTES;
        if (kt + 1 < nstage)
            stage_async(kt + 1, lds_base + (uint32_t)(((kt + 1) & 1) * BUF_BYTES));

#pragma unroll
        for (int ntp = 0; ntp < KT / 32; ++ntp) {
            // 4 independent accumulator chains: {M0,M1} x {N-tile0, N-tile1}
            v8f accA0 = {0.f, 0.f, 0.f, 0.f, 0.f, 0.f, 0.f, 0.f};
            v8f accB0 = {0.f, 0.f, 0.f, 0.f, 0.f, 0.f, 0.f, 0.f};
            v8f accA1 = {0.f, 0.f, 0.f, 0.f, 0.f, 0.f, 0.f, 0.f};
            v8f accB1 = {0.f, 0.f, 0.f, 0.f, 0.f, 0.f, 0.f, 0.f};
            // B 32x16 layout: lane holds column N=nl; element e -> K = (e&7)+8h+16*(e>>3)
            // => two 16B chunks per fragment at col offsets cb+8h and cb+16+8h.
            const char* brow0 = cur + (size_t)((2 * ntp + 0) * 16 + nl) * LDS_ROW + 16 * h;
            const char* brow1 = cur + (size_t)((2 * ntp + 1) * 16 + nl) * LDS_ROW + 16 * h;
#pragma unroll
            for (int cc = 0; cc < 16; ++cc) {
                const v8bf* p0 = (const v8bf*)(brow0 + cc * 64);
                const v8bf* p1 = (const v8bf*)(brow1 + cc * 64);
                v8bf f0lo = p0[0], f0hi = p0[2];
                v8bf f1lo = p1[0], f1hi = p1[2];
                v16bf f0 = __builtin_shufflevector(f0lo, f0hi, 0, 1, 2, 3, 4, 5, 6, 7,
                                                   8, 9, 10, 11, 12, 13, 14, 15);
                v16bf f1 = __builtin_shufflevector(f1lo, f1hi, 0, 1, 2, 3, 4, 5, 6, 7,
                                                   8, 9, 10, 11, 12, 13, 14, 15);
                accA0 = __builtin_amdgcn_wmma_f32_16x16x32_bf16(
                    false, a0[cc], false, f0, (short)0, accA0, false, false);
                accB0 = __builtin_amdgcn_wmma_f32_16x16x32_bf16(
                    false, a1[cc], false, f0, (short)0, accB0, false, false);
                accA1 = __builtin_amdgcn_wmma_f32_16x16x32_bf16(
                    false, a0[cc], false, f1, (short)0, accA1, false, false);
                accB1 = __builtin_amdgcn_wmma_f32_16x16x32_bf16(
                    false, a1[cc], false, f1, (short)0, accB1, false, false);
            }
            const int code0 = kt * KT + (2 * ntp + 0) * 16 + nl;
            const int code1 = code0 + 16;
            const float en0 = enorm[code0];
            const float en1 = enorm[code1];
#pragma unroll
            for (int i = 0; i < 8; ++i) {
                float sA0 = en0 - 2.0f * accA0[i];
                if (sA0 < mind0[i]) { mind0[i] = sA0; midx0[i] = code0; }
                float sB0 = en0 - 2.0f * accB0[i];
                if (sB0 < mind1[i]) { mind1[i] = sB0; midx1[i] = code0; }
                float sA1 = en1 - 2.0f * accA1[i];
                if (sA1 < mind0[i]) { mind0[i] = sA1; midx0[i] = code1; }
                float sB1 = en1 - 2.0f * accB1[i];
                if (sB1 < mind1[i]) { mind1[i] = sB1; midx1[i] = code1; }
            }
        }
        asm volatile("s_wait_asynccnt 0x0" ::: "memory");
        __syncthreads();
    }

    // Reduce min across the 16 lanes of each half (N dimension), both subtiles.
#pragma unroll
    for (int i = 0; i < 8; ++i) {
#pragma unroll
        for (int off = 1; off < 16; off <<= 1) {
            float om0 = __shfl_xor(mind0[i], off, 32);
            int   oi0 = __shfl_xor(midx0[i], off, 32);
            if (om0 < mind0[i] || (om0 == mind0[i] && oi0 < midx0[i])) {
                mind0[i] = om0; midx0[i] = oi0;
            }
            float om1 = __shfl_xor(mind1[i], off, 32);
            int   oi1 = __shfl_xor(midx1[i], off, 32);
            if (om1 < mind1[i] || (om1 == mind1[i] && oi1 < midx1[i])) {
                mind1[i] = om1; midx1[i] = oi1;
            }
        }
    }
    if (nl == 0) {
#pragma unroll
        for (int i = 0; i < 8; ++i) {
            idx_out[row_base + i + 8 * h]      = midx0[i];  // D-layout: M = i + 8h
            idx_out[row_base + 16 + i + 8 * h] = midx1[i];
        }
    }
}

// ---------------------------------------------------------------------------
// Gather z_q = emb[idx] back to [B,C,F,H,W] + MSE partial reduction
// ---------------------------------------------------------------------------
__global__ void k_gather(const float* __restrict__ emb, const float* __restrict__ z,
                         const int* __restrict__ idx, float* __restrict__ zq,
                         float* __restrict__ acc) {
    __shared__ float tile[32][33];
    __shared__ float red[256];
    const int s0 = blockIdx.x * 32;
    const int c0 = blockIdx.y * 32;
    const int b  = blockIdx.z;
    const int tx = threadIdx.x, ty = threadIdx.y;
#pragma unroll
    for (int j = 0; j < 4; ++j) {
        int sl = ty + 8 * j;
        int k  = idx[b * SPATIAL + s0 + sl];
        tile[tx][sl] = emb[(size_t)k * EMB_DIM + c0 + tx];  // coalesced in c
    }
    __syncthreads();
    float lsum = 0.f;
#pragma unroll
    for (int j = 0; j < 4; ++j) {
        int cl = ty + 8 * j;
        size_t o = (size_t)b * EMB_DIM * SPATIAL + (size_t)(c0 + cl) * SPATIAL + s0 + tx;
        float v = tile[cl][tx];
        zq[o] = v;                                          // coalesced in s
        float d = v - z[o];
        lsum += d * d;
    }
    const int t = ty * 32 + tx;
    red[t] = lsum;
    __syncthreads();
    for (int off = 128; off > 0; off >>= 1) {
        if (t < off) red[t] += red[t + off];
        __syncthreads();
    }
    if (t == 0) atomicAdd(acc, red[0]);
}

__global__ void k_final(const float* __restrict__ acc, float* __restrict__ loss_out) {
    float mse = acc[0] / (float)((size_t)BATCH * EMB_DIM * SPATIAL);
    loss_out[0] = mse + 0.25f * mse;   // codebook + commitment (numerically equal)
}

// ---------------------------------------------------------------------------
extern "C" void kernel_launch(void* const* d_in, const int* in_sizes, int n_in,
                              void* d_out, int out_size, void* d_ws, size_t ws_size,
                              hipStream_t stream) {
    const float* z   = (const float*)d_in[0];
    const float* emb = (const float*)d_in[1];

    float* out     = (float*)d_out;
    float* zq      = out;                                         // 8388608 floats
    float* loss    = out + (size_t)BATCH * EMB_DIM * SPATIAL;     // 1 float
    int*   indices = (int*)(loss + 1);                            // 16384 int32

    char*   ws    = (char*)d_ws;
    bf16_t* zbf   = (bf16_t*)ws;                                  // 16 MB
    bf16_t* ebf   = (bf16_t*)(ws + (size_t)NVEC * EMB_DIM * 2);   // 8 MB
    float*  enorm = (float*)(ws + (size_t)NVEC * EMB_DIM * 2
                                + (size_t)NUM_EMB * EMB_DIM * 2); // 32 KB
    float*  acc   = enorm + NUM_EMB;                              // 4 B

    hipMemsetAsync(acc, 0, sizeof(float), stream);

    dim3 t32x8(32, 8);
    k_zprep<<<dim3(SPATIAL / 32, EMB_DIM / 32, BATCH), t32x8, 0, stream>>>(z, zbf);
    k_eprep<<<dim3(NUM_EMB), dim3(256), 0, stream>>>(emb, ebf, enorm);

    size_t smem = 2 * BUF_BYTES;                                  // 133,120 B (<320KB WGP)
    k_argmin<<<dim3(NVEC / 256), dim3(256), smem, stream>>>(zbf, ebf, enorm, indices);

    k_gather<<<dim3(SPATIAL / 32, EMB_DIM / 32, BATCH), t32x8, 0, stream>>>(emb, z, indices, zq, acc);
    k_final<<<1, 1, 0, stream>>>(acc, loss);
}
